// metapath_product_average_attention_35158602285300
// MI455X (gfx1250) — matrix operand order repaired and checked by reading, not compiled, not run on hardware
//
#include <hip/hip_runtime.h>

typedef __attribute__((ext_vector_type(16))) _Float16 v16h;
typedef __attribute__((ext_vector_type(8)))  float    v8f;
typedef __attribute__((ext_vector_type(4)))  float    v4f;

#if __has_builtin(__builtin_amdgcn_tanhf)
__device__ __forceinline__ float fast_tanh(float x) { return __builtin_amdgcn_tanhf(x); }
#else
__device__ __forceinline__ float fast_tanh(float x) { return tanhf(x); }
#endif

// ---------------------------------------------------------------------------
// Tile body: 16 flat rows -> A(16x128 f16) x W(128x16 f16) -> C(16x16 f32),
// then acc4[m] += tanh(C[r][p]+bias[p]) * mpref[r&3][p] (per-lane partials).
// GUARD=true adds per-row bounds checks (tail tile only).
// ---------------------------------------------------------------------------
template <bool GUARD>
__device__ __forceinline__
void process_tile(long long base, int p, int hi,
                  const float* __restrict__ tables,
                  const long long* __restrict__ indices,
                  const v16h* btile, float bp, const float* mp,
                  long long N, long long totalRows, float* acc4) {
    // Row assignment for A-matrix: lane (and lane+16) own row laneM = p.
    long long flat = base + p;
    if (GUARD && flat >= totalRows) flat = totalRows - 1;     // tail clamp
    const int m   = (int)(flat & 3);
    long long row = indices[flat];
    if (row < 0) row = 0; else if (row >= N) row = N - 1;     // safety clamp
    const float* arow = tables + ((long long)m * N + row) * 128;

    v8f c = {0.f, 0.f, 0.f, 0.f, 0.f, 0.f, 0.f, 0.f};
#pragma unroll
    for (int ks = 0; ks < 4; ++ks) {
        // A layout (16-bit 16x32): halves 0-7 = K k0+hi*8.., halves 8-15 = K k0+16+hi*8..
        const float* pa = arow + ks * 32 + hi * 8;
        const v4f f0 = *(const v4f*)(pa);
        const v4f f1 = *(const v4f*)(pa + 4);
        const v4f f2 = *(const v4f*)(pa + 16);
        const v4f f3 = *(const v4f*)(pa + 20);
        v16h a;
#pragma unroll
        for (int j = 0; j < 4; ++j) {
            a[j]      = (_Float16)f0[j];
            a[4 + j]  = (_Float16)f1[j];
            a[8 + j]  = (_Float16)f2[j];
            a[12 + j] = (_Float16)f3[j];
        }
        c = __builtin_amdgcn_wmma_f32_16x16x32_f16(
                false, a, false, btile[ks], (short)0, c, false, false);
    }

    // C element j on this lane = row r = j + 8*hi, column p; r&3 == j&3.
#pragma unroll
    for (int j = 0; j < 8; ++j) {
        if (!GUARD || (base + j + 8 * hi < totalRows))
            acc4[j & 3] = fmaf(fast_tanh(c[j] + bp), mp[j & 3], acc4[j & 3]);
    }
}

// ---------------------------------------------------------------------------
// Kernel 1: per-(b,m) attention logits via WMMA, reduced to 4 per-m sums.
// Main loop runs only over provably-full tiles (no per-iteration bounds
// branch); at most one tail tile is handled once by a single wave.
// ---------------------------------------------------------------------------
__global__ __launch_bounds__(256)
void attn_weights_kernel(const float* __restrict__ tables,
                         const long long* __restrict__ indices,
                         const float* __restrict__ W,
                         const float* __restrict__ bias,
                         const float* __restrict__ mpref,
                         float* __restrict__ acc,      // 4 floats, pre-zeroed
                         long long N, int numFullTiles, long long totalRows) {
    const int lane  = threadIdx.x & 31;
    const int p     = lane & 15;          // C/D & B column this lane owns
    const int hi    = lane >> 4;          // half-wave selector
    const int wave  = blockIdx.x * (blockDim.x >> 5) + (threadIdx.x >> 5);
    const int nWave = gridDim.x * (blockDim.x >> 5);

    // Preload W as 4 f16 B-tiles (32x16 each). B layout: lane holds column p,
    // halves 0-7 = K kbase+hi*8.., halves 8-15 = K kbase+16+hi*8..
    v16h btile[4];
#pragma unroll
    for (int ks = 0; ks < 4; ++ks) {
        const int kbase = ks * 32 + hi * 8;
#pragma unroll
        for (int j = 0; j < 16; ++j)
            btile[ks][j] = (_Float16)W[(kbase + j + (j >= 8 ? 8 : 0)) * 16 + p];
    }
    const float bp = bias[p];
    float mp[4];
#pragma unroll
    for (int mm = 0; mm < 4; ++mm) mp[mm] = mpref[mm * 16 + p];

    float acc4[4] = {0.f, 0.f, 0.f, 0.f};

    // Steady state: full tiles, unconditional accumulation.
    for (int tile = wave; tile < numFullTiles; tile += nWave)
        process_tile<false>((long long)tile * 16, p, hi, tables, indices,
                            btile, bp, mp, N, totalRows, acc4);

    // Tail tile (totalRows % 16 rows), processed by exactly one wave.
    if (((long long)numFullTiles * 16 < totalRows) &&
        (wave == (numFullTiles % nWave)))
        process_tile<true>((long long)numFullTiles * 16, p, hi, tables, indices,
                           btile, bp, mp, N, totalRows, acc4);

    // One-time 32-lane butterfly reduction of the 4 per-m partials.
#pragma unroll
    for (int mm = 0; mm < 4; ++mm) {
#pragma unroll
        for (int d = 1; d < 32; d <<= 1)
            acc4[mm] += __shfl_xor(acc4[mm], d, 32);
    }
    if (lane == 0) {
#pragma unroll
        for (int mm = 0; mm < 4; ++mm)
            atomicAdd(&acc[mm], acc4[mm]);
    }
}

// ---------------------------------------------------------------------------
// Kernel 2: softmax over the 4 per-m means.
// ---------------------------------------------------------------------------
__global__ void softmax4_kernel(const float* __restrict__ acc,
                                float* __restrict__ wts, float invB) {
    if (threadIdx.x == 0 && blockIdx.x == 0) {
        float v[4];
        float mx = -1e30f;
        for (int i = 0; i < 4; ++i) { v[i] = acc[i] * invB; mx = fmaxf(mx, v[i]); }
        float s = 0.f;
        for (int i = 0; i < 4; ++i) { v[i] = __expf(v[i] - mx); s += v[i]; }
        const float inv = 1.f / s;
        for (int i = 0; i < 4; ++i) wts[i] = v[i] * inv;
    }
}

// ---------------------------------------------------------------------------
// Kernel 3: out[b][:] = sum_m wts[m] * tables[m][idx[b][m]][:]
// One wave per b; each lane owns 4 consecutive d values (b128 load/store).
// ---------------------------------------------------------------------------
__global__ __launch_bounds__(256)
void weighted_sum_kernel(const float* __restrict__ tables,
                         const long long* __restrict__ indices,
                         const float* __restrict__ wts,
                         float* __restrict__ out,
                         long long N, int B) {
    const int lane  = threadIdx.x & 31;
    const int wave  = blockIdx.x * (blockDim.x >> 5) + (threadIdx.x >> 5);
    const int nWave = gridDim.x * (blockDim.x >> 5);

    const float w0 = wts[0], w1 = wts[1], w2 = wts[2], w3 = wts[3];

    for (int b = wave; b < B; b += nWave) {
        const long long* ib = indices + (long long)b * 4;
        long long i0 = ib[0], i1 = ib[1], i2 = ib[2], i3 = ib[3];
        if (i0 < 0) i0 = 0; else if (i0 >= N) i0 = N - 1;
        if (i1 < 0) i1 = 0; else if (i1 >= N) i1 = N - 1;
        if (i2 < 0) i2 = 0; else if (i2 >= N) i2 = N - 1;
        if (i3 < 0) i3 = 0; else if (i3 >= N) i3 = N - 1;

        const int d = lane * 4;
        const v4f r0 = *(const v4f*)(tables + (0LL * N + i0) * 128 + d);
        const v4f r1 = *(const v4f*)(tables + (1LL * N + i1) * 128 + d);
        const v4f r2 = *(const v4f*)(tables + (2LL * N + i2) * 128 + d);
        const v4f r3 = *(const v4f*)(tables + (3LL * N + i3) * 128 + d);

        v4f s;
#pragma unroll
        for (int j = 0; j < 4; ++j)
            s[j] = fmaf(w3, r3[j], fmaf(w2, r2[j], fmaf(w1, r1[j], w0 * r0[j])));

        *(v4f*)(out + (long long)b * 128 + d) = s;
    }
}

// ---------------------------------------------------------------------------
extern "C" void kernel_launch(void* const* d_in, const int* in_sizes, int n_in,
                              void* d_out, int out_size, void* d_ws, size_t ws_size,
                              hipStream_t stream) {
    const float*     tables = (const float*)d_in[0];      // (M, N, D) f32
    const long long* idx    = (const long long*)d_in[1];  // (B, M) int64
    const float*     W      = (const float*)d_in[2];      // (D, P) f32
    const float*     bias   = (const float*)d_in[3];      // (P,)   f32
    const float*     mpref  = (const float*)d_in[4];      // (M, P) f32
    float*           out    = (float*)d_out;              // (B, D) f32

    const int       P            = in_sizes[3];                              // 16
    const int       M            = in_sizes[4] / P;                          // 4
    const int       D            = in_sizes[2] / P;                          // 128
    const long long B            = (long long)in_sizes[1] / M;               // 200000
    const long long N            = (long long)in_sizes[0] / ((long long)M * D); // 300000
    const long long totalRows    = B * M;
    const int       numFullTiles = (int)(totalRows / 16);
    (void)D; (void)n_in; (void)out_size; (void)ws_size;

    float* acc = (float*)d_ws;   // 4 floats: per-m logit sums
    float* wts = acc + 4;        // 4 floats: softmax weights

    hipMemsetAsync(acc, 0, 4 * sizeof(float), stream);

    attn_weights_kernel<<<2048, 256, 0, stream>>>(tables, idx, W, bias, mpref,
                                                  acc, N, numFullTiles, totalRows);
    softmax4_kernel<<<1, 32, 0, stream>>>(acc, wts, 1.0f / (float)B);
    weighted_sum_kernel<<<4096, 256, 0, stream>>>(tables, idx, wts, out,
                                                  N, (int)B);
}
